// Transformer_34849364639910
// MI455X (gfx1250) — compile-verified, matrix operand order checked by reading, not run on hardware
//
#include <hip/hip_runtime.h>
#include <hip/hip_bf16.h>
#include <stdint.h>

#define BB 4
#define SS 2048
#define CC 64
#define HH 8
#define HCC 512
#define MM (BB * SS)
#define NEGV -1.0e9f

typedef __attribute__((ext_vector_type(16))) __bf16 v16bf;
typedef __attribute__((ext_vector_type(8)))  float  v8f;
typedef int vi4 __attribute__((vector_size(16)));
typedef __attribute__((address_space(1))) vi4* gvi4p;   // global int4*
typedef __attribute__((address_space(3))) vi4* lvi4p;   // LDS int4*

union Frag16 { uint32_t u[8]; v16bf v; };

#ifndef __has_builtin
#define __has_builtin(x) 0
#endif
#if __has_builtin(__builtin_amdgcn_global_load_async_to_lds_b128) && \
    __has_builtin(__builtin_amdgcn_s_wait_asynccnt)
#define HAVE_ASYNC 1
#else
#define HAVE_ASYNC 0
#endif

// Copy 16B global -> LDS. Async (ASYNCcnt-tracked) when available.
__device__ __forceinline__ void stage16(unsigned short* ldst,
                                        const unsigned short* gsrc) {
#if HAVE_ASYNC
  __builtin_amdgcn_global_load_async_to_lds_b128(
      (gvi4p)(void*)gsrc, (lvi4p)(void*)ldst, 0, 0);
#else
  *(uint4*)ldst = *(const uint4*)gsrc;
#endif
}
#if HAVE_ASYNC
#define STAGE_WAIT(n) __builtin_amdgcn_s_wait_asynccnt(n)
#else
#define STAGE_WAIT(n) ((void)0)
#endif

// 16-bit A/B fragment: dwords 0-3 = K hf*8..hf*8+7 (contiguous),
// dwords 4-7 = K 16+hf*8..+7 (contiguous) -> two 16-byte loads.
__device__ __forceinline__ void load_frag(Frag16& fr, const unsigned short* base) {
  uint4 lo = *(const uint4*)(base);
  uint4 hi = *(const uint4*)(base + 16);
  fr.u[0] = lo.x; fr.u[1] = lo.y; fr.u[2] = lo.z; fr.u[3] = lo.w;
  fr.u[4] = hi.x; fr.u[5] = hi.y; fr.u[6] = hi.z; fr.u[7] = hi.w;
}

__device__ __forceinline__ unsigned short bf16_1(float f) {
  uint32_t a = __float_as_uint(f);
  return (unsigned short)((a + 0x7FFFu + ((a >> 16) & 1u)) >> 16);
}

__device__ __forceinline__ float hredu_max(float v) {
#pragma unroll
  for (int m = 1; m < 16; m <<= 1) v = fmaxf(v, __shfl_xor(v, m, 32));
  return v;
}
__device__ __forceinline__ float hredu_sum(float v) {
#pragma unroll
  for (int m = 1; m < 16; m <<= 1) v += __shfl_xor(v, m, 32);
  return v;
}

// ---------------------------------------------------------------------------
// Kernel 0: one-shot fp32 -> bf16 conversion (+ weight transposes).
// ---------------------------------------------------------------------------
#define NXE (MM * CC)            // 524288
#define NWE (HCC * CC)           // 32768
__global__ __launch_bounds__(256)
void convert_kernel(const float* __restrict__ query,
                    const float* __restrict__ content,
                    const float* __restrict__ Wq, const float* __restrict__ Wk,
                    const float* __restrict__ Wv, const float* __restrict__ Wo,
                    unsigned short* __restrict__ qxb,
                    unsigned short* __restrict__ cxb,
                    unsigned short* __restrict__ WqT,
                    unsigned short* __restrict__ WkT,
                    unsigned short* __restrict__ WvT,
                    unsigned short* __restrict__ WoT) {
  int tid = blockIdx.x * blockDim.x + threadIdx.x;
  if (tid < NXE) {
    qxb[tid] = bf16_1(query[tid]);
    cxb[tid] = bf16_1(content[tid]);
  }
  if (tid < NWE) {
    int n = tid / CC, k = tid % CC;          // QKV weights: [C,HCC] -> [HCC,C]
    WqT[tid] = bf16_1(Wq[(size_t)k * HCC + n]);
    WkT[tid] = bf16_1(Wk[(size_t)k * HCC + n]);
    WvT[tid] = bf16_1(Wv[(size_t)k * HCC + n]);
    int n2 = tid / HCC, k2 = tid % HCC;      // Wo: [HCC,C] -> [C,HCC]
    WoT[tid] = bf16_1(Wo[(size_t)k2 * CC + n2]);
  }
}

// ---------------------------------------------------------------------------
// Kernel 1: QKV projection. One wave = one 16x16 tile of [8192,64]@[64,512].
// q,k stored bf16 [B,H,S,C]; v stored bf16 transposed [B,H,C,S].
// ---------------------------------------------------------------------------
__global__ __launch_bounds__(32)
void qkv_proj_kernel(const unsigned short* __restrict__ qxb,
                     const unsigned short* __restrict__ cxb,
                     const unsigned short* __restrict__ WqT,
                     const unsigned short* __restrict__ WkT,
                     const unsigned short* __restrict__ WvT,
                     const float* __restrict__ bq,
                     const float* __restrict__ bk,
                     const float* __restrict__ bv,
                     unsigned short* __restrict__ qb,
                     unsigned short* __restrict__ kb,
                     unsigned short* __restrict__ vt) {
  const int lane = threadIdx.x & 31;
  const int hf = lane >> 4;
  const int ln = lane & 15;
  const int row0 = blockIdx.x * 16;
  const int col0 = blockIdx.y * 16;
  const int which = blockIdx.z;

  const unsigned short* X  = (which == 0) ? qxb : cxb;
  const unsigned short* WT = (which == 0) ? WqT : ((which == 1) ? WkT : WvT);
  const float* bias        = (which == 0) ? bq  : ((which == 1) ? bk  : bv);

  const unsigned short* xrow = X + (size_t)(row0 + ln) * CC + hf * 8;
  const unsigned short* wrow = WT + (size_t)(col0 + ln) * CC + hf * 8;

  Frag16 a0, a1, b0, b1;
  load_frag(a0, xrow);       load_frag(a1, xrow + 32);
  load_frag(b0, wrow);       load_frag(b1, wrow + 32);

  v8f c = {};
  c = __builtin_amdgcn_wmma_f32_16x16x32_bf16(false, a0.v, false, b0.v,
                                              (short)0, c, false, false);
  c = __builtin_amdgcn_wmma_f32_16x16x32_bf16(false, a1.v, false, b1.v,
                                              (short)0, c, false, false);

  const float bval = bias[col0 + ln];
  const int col = col0 + ln;
  const int h = col >> 6;
  const int ch = col & 63;
  const int bi = row0 / SS;                 // 16-row tile never straddles batch
  const int s0 = (row0 % SS) + 8 * hf;

  unsigned short* dst;
  size_t idx0;
  int stride;
  if (which == 0) {
    dst = qb;  idx0 = (((size_t)bi * HH + h) * SS + s0) * CC + ch;  stride = CC;
  } else if (which == 1) {
    dst = kb;  idx0 = (((size_t)bi * HH + h) * SS + s0) * CC + ch;  stride = CC;
  } else {
    dst = vt;  idx0 = (((size_t)bi * HH + h) * CC + ch) * SS + s0;  stride = 1;
  }
#pragma unroll
  for (int r = 0; r < 8; ++r)
    dst[idx0 + (size_t)r * stride] = bf16_1(c[r] + bval);
}

// ---------------------------------------------------------------------------
// Kernel 2: causal flash attention with double-buffered async K/V staging
// in LDS. One wave per (b,h, 16-row q tile).
// ---------------------------------------------------------------------------
__global__ __launch_bounds__(32)
void flash_attn_kernel(const unsigned short* __restrict__ qb,
                       const unsigned short* __restrict__ kb,
                       const unsigned short* __restrict__ vt,
                       unsigned short* __restrict__ att) {
  __shared__ unsigned short kbuf[2][32 * CC];   // 2 x 4KB: 32 keys x 64 chans
  __shared__ unsigned short vbuf[2][CC * 32];   // 2 x 4KB: 64 chans x 32 keys
  __shared__ unsigned short plds[16 * 32];      // 1KB: P tile (bf16)

  const int lane = threadIdx.x & 31;
  const int hf = lane >> 4;
  const int ln = lane & 15;
  const int q0 = blockIdx.x * 16;
  const int bh = blockIdx.y;
  const int bi = bh / HH, h = bh % HH;

  const unsigned short* qp = qb + (size_t)bh * SS * CC;
  const unsigned short* kp = kb + (size_t)bh * SS * CC;
  const unsigned short* vp = vt + (size_t)bh * CC * SS;

  // Stage one 32-key block (K: contiguous 4KB; V: 64 rows x 64B) into LDS.
  // 16 x 16B-per-lane copies = 16 ASYNCcnt ops per block.
  auto issue = [&](int j, int b) {
#pragma unroll
    for (int i = 0; i < 8; ++i) {
      int e = i * 256 + lane * 8;
      stage16(&kbuf[b][e], kp + (size_t)j * 32 * CC + e);
    }
#pragma unroll
    for (int i = 0; i < 8; ++i) {
      int chunk = i * 32 + lane;            // 256 chunks of 16B
      int c = chunk >> 2, e0 = (chunk & 3) * 8;
      stage16(&vbuf[b][c * 32 + e0], vp + (size_t)c * SS + j * 32 + e0);
    }
  };

  // Persistent q-tile A fragments (16 x 64).
  Frag16 aq0, aq1;
  {
    const unsigned short* qrow = qp + (size_t)(q0 + ln) * CC + hf * 8;
    load_frag(aq0, qrow);
    load_frag(aq1, qrow + 32);
  }

  v8f o[4] = {{}, {}, {}, {}};
  float rmax[8], rsum[8];
#pragma unroll
  for (int r = 0; r < 8; ++r) { rmax[r] = -__builtin_inff(); rsum[r] = 0.f; }

  const int nkb = (q0 + 15) / 32 + 1;        // causal: blocks <= diagonal
  issue(0, 0);

  for (int kbk = 0; kbk < nkb; ++kbk) {
    const int buf = kbk & 1;
    const bool more = (kbk + 1 < nkb);
    if (more) issue(kbk + 1, buf ^ 1);       // overlap copy with compute
    if (more) STAGE_WAIT(16);                // current block's 16 ops retired
    else      STAGE_WAIT(0);

    // Deep prefetch (kbk+2) to warm L2 lines for the async engine.
    if (kbk + 2 < nkb) {
      __builtin_prefetch(kp + (size_t)(kbk + 2) * 32 * CC + lane * 64, 0, 3);
      __builtin_prefetch(vp + (size_t)(2 * lane) * SS + (kbk + 2) * 32, 0, 3);
      __builtin_prefetch(vp + (size_t)(2 * lane + 1) * SS + (kbk + 2) * 32, 0, 3);
    }

    // --- scores: two 16x16 tiles covering 32 keys (K frags via ds_load) ---
    v8f st[2];
#pragma unroll
    for (int t = 0; t < 2; ++t) {
      const unsigned short* krow = &kbuf[buf][(t * 16 + ln) * CC + hf * 8];
      Frag16 bk0, bk1;
      load_frag(bk0, krow);
      load_frag(bk1, krow + 32);
      v8f s = {};
      s = __builtin_amdgcn_wmma_f32_16x16x32_bf16(false, aq0.v, false, bk0.v,
                                                  (short)0, s, false, false);
      s = __builtin_amdgcn_wmma_f32_16x16x32_bf16(false, aq1.v, false, bk1.v,
                                                  (short)0, s, false, false);
      int col = kbk * 32 + t * 16 + ln;
#pragma unroll
      for (int r = 0; r < 8; ++r) {
        int qrow = q0 + r + 8 * hf;
        if (col > qrow) s[r] = NEGV;         // causal mask
      }
      st[t] = s;
    }

    // --- online softmax (row stats live in 16-lane halves) ---
    float p0[8], p1[8], scale[8];
#pragma unroll
    for (int r = 0; r < 8; ++r) {
      float tmax = hredu_max(fmaxf(st[0][r], st[1][r]));
      float mnew = fmaxf(rmax[r], tmax);
      scale[r] = __expf(rmax[r] - mnew);     // 0 when rmax was -inf
      p0[r] = __expf(st[0][r] - mnew);
      p1[r] = __expf(st[1][r] - mnew);
      float ls = hredu_sum(p0[r] + p1[r]);
      rsum[r] = rsum[r] * scale[r] + ls;
      rmax[r] = mnew;
    }
#pragma unroll
    for (int r = 0; r < 8; ++r) {
      o[0][r] *= scale[r]; o[1][r] *= scale[r];
      o[2][r] *= scale[r]; o[3][r] *= scale[r];
    }

    // --- stage P (bf16) in LDS, reread in A-fragment layout ---
    __syncthreads();
#pragma unroll
    for (int r = 0; r < 8; ++r) {
      int row = r + 8 * hf;
      plds[row * 32 + ln]      = bf16_1(p0[r]);
      plds[row * 32 + 16 + ln] = bf16_1(p1[r]);
    }
    __syncthreads();
    Frag16 ap;
    load_frag(ap, plds + ln * 32 + hf * 8);  // ds_load_b128 x2

    // --- P @ V: 4 channel tiles (V frags via ds_load from staged tile) ---
#pragma unroll
    for (int cb = 0; cb < 4; ++cb) {
      Frag16 bv2;
      load_frag(bv2, &vbuf[buf][(cb * 16 + ln) * 32 + hf * 8]);
      o[cb] = __builtin_amdgcn_wmma_f32_16x16x32_bf16(false, ap.v, false, bv2.v,
                                                      (short)0, o[cb], false, false);
    }
  }

  // --- normalize + store bf16 to att[b][s][h*64 + c] ---
#pragma unroll
  for (int r = 0; r < 8; ++r) {
    float inv = 1.0f / rsum[r];
    int s = q0 + r + 8 * hf;
    size_t base = ((size_t)bi * SS + s) * HCC + (size_t)h * CC;
    att[base + 0 * 16 + ln] = bf16_1(o[0][r] * inv);
    att[base + 1 * 16 + ln] = bf16_1(o[1][r] * inv);
    att[base + 2 * 16 + ln] = bf16_1(o[2][r] * inv);
    att[base + 3 * 16 + ln] = bf16_1(o[3][r] * inv);
  }
}

// ---------------------------------------------------------------------------
// Kernel 3: output projection [8192,512] @ [512,64] + bo -> fp32 out.
// ---------------------------------------------------------------------------
__global__ __launch_bounds__(32)
void out_proj_kernel(const unsigned short* __restrict__ att,
                     const unsigned short* __restrict__ WoT,
                     const float* __restrict__ bo,
                     float* __restrict__ out) {
  const int lane = threadIdx.x & 31;
  const int hf = lane >> 4, ln = lane & 15;
  const int row0 = blockIdx.x * 16;
  const int col0 = blockIdx.y * 16;

  const unsigned short* arow = att + (size_t)(row0 + ln) * HCC + hf * 8;
  const unsigned short* wrow = WoT + (size_t)(col0 + ln) * HCC + hf * 8;

  v8f c = {};
  for (int kb2 = 0; kb2 < HCC / 32; ++kb2) {
    Frag16 a, bw;
    load_frag(a,  arow + kb2 * 32);
    load_frag(bw, wrow + kb2 * 32);
    c = __builtin_amdgcn_wmma_f32_16x16x32_bf16(false, a.v, false, bw.v,
                                                (short)0, c, false, false);
  }
  float bval = bo[col0 + ln];
#pragma unroll
  for (int r = 0; r < 8; ++r) {
    int row = row0 + r + 8 * hf;
    out[(size_t)row * CC + col0 + ln] = c[r] + bval;
  }
}

// ---------------------------------------------------------------------------
extern "C" void kernel_launch(void* const* d_in, const int* in_sizes, int n_in,
                              void* d_out, int out_size, void* d_ws, size_t ws_size,
                              hipStream_t stream) {
  const float* query   = (const float*)d_in[0];
  const float* content = (const float*)d_in[1];
  const float* Wq = (const float*)d_in[2];
  const float* bq = (const float*)d_in[3];
  const float* Wk = (const float*)d_in[4];
  const float* bk = (const float*)d_in[5];
  const float* Wv = (const float*)d_in[6];
  const float* bv = (const float*)d_in[7];
  const float* Wo = (const float*)d_in[8];
  const float* bo = (const float*)d_in[9];
  float* out = (float*)d_out;

  const size_t BHSC = (size_t)BB * HH * SS * CC;   // 4,194,304
  unsigned short* p = (unsigned short*)d_ws;
  unsigned short* qxb = p;            p += NXE;     // bf16 query  [M,C]
  unsigned short* cxb = p;            p += NXE;     // bf16 content[M,C]
  unsigned short* WqT = p;            p += NWE;     // bf16 [HCC,C]
  unsigned short* WkT = p;            p += NWE;
  unsigned short* WvT = p;            p += NWE;
  unsigned short* WoT = p;            p += NWE;     // bf16 [C,HCC]
  unsigned short* qb  = p;            p += BHSC;    // [B,H,S,C]
  unsigned short* kb  = p;            p += BHSC;    // [B,H,S,C]
  unsigned short* vt  = p;            p += BHSC;    // [B,H,C,S]
  unsigned short* att = p;                          // [B,S,HC]

  convert_kernel<<<(NXE + 255) / 256, 256, 0, stream>>>(
      query, content, Wq, Wk, Wv, Wo, qxb, cxb, WqT, WkT, WvT, WoT);

  dim3 g1(MM / 16, HCC / 16, 3);
  qkv_proj_kernel<<<g1, 32, 0, stream>>>(qxb, cxb, WqT, WkT, WvT,
                                         bq, bk, bv, qb, kb, vt);
  dim3 g2(SS / 16, BB * HH);
  flash_attn_kernel<<<g2, 32, 0, stream>>>(qb, kb, vt, att);
  dim3 g3(MM / 16, CC / 16);
  out_proj_kernel<<<g3, 32, 0, stream>>>(att, WoT, bo, out);
}
